// ProgressiveModel_82832739270887
// MI455X (gfx1250) — matrix operand order repaired and checked by reading, not compiled, not run on hardware
//
#include <hip/hip_runtime.h>

typedef __attribute__((ext_vector_type(16))) _Float16 v16h;
typedef __attribute__((ext_vector_type(8)))  _Float16 v8h;
typedef __attribute__((ext_vector_type(8)))  float    v8f;
typedef __attribute__((ext_vector_type(4)))  unsigned int u32x4;
typedef __attribute__((ext_vector_type(8)))  unsigned int u32x8;

#define D_MODEL 1024
#define N_REG   8
#define D_REG   32
#define BATCH   8
#define SEQL    2048
#define NTOK    (BATCH * SEQL)   // 16384
#define LN_EPS  1e-5f
#define KCHUNK  128              // K elements staged per TDM transfer

// ---------------------------------------------------------------------------
// Kernel 0: stage weights as f16.
// Wcat (64 x 1024): rows 0-7 = Wrq, 8-15 = Wwq, 16-47 = Wwv, 48 = Wwg, 49-63 = 0
// Wrp16 (1024 x 32): straight f32->f16 copy of Wrp.
// ---------------------------------------------------------------------------
__global__ void prep_weights(const float* __restrict__ Wrq, const float* __restrict__ Wwq,
                             const float* __restrict__ Wwv, const float* __restrict__ Wwg,
                             const float* __restrict__ Wrp,
                             _Float16* __restrict__ Wcat, _Float16* __restrict__ Wrp16) {
    int i = blockIdx.x * blockDim.x + threadIdx.x;
    const int total_cat = 64 * D_MODEL;
    if (i < total_cat) {
        int n = i / D_MODEL, k = i % D_MODEL;
        float v = 0.0f;
        if (n < 8)       v = Wrq[n * D_MODEL + k];
        else if (n < 16) v = Wwq[(n - 8) * D_MODEL + k];
        else if (n < 48) v = Wwv[(n - 16) * D_MODEL + k];
        else if (n == 48) v = Wwg[k];
        Wcat[i] = (_Float16)v;
    } else {
        int j = i - total_cat;
        if (j < D_MODEL * D_REG) Wrp16[j] = (_Float16)Wrp[j];
    }
}

// ---------------------------------------------------------------------------
// Kernel 1: LayerNorm, f32 in -> f16 out. One block (256 thr) per token row.
// ---------------------------------------------------------------------------
__global__ __launch_bounds__(256) void ln_kernel(const float* __restrict__ x,
                                                 const float* __restrict__ w,
                                                 const float* __restrict__ b,
                                                 _Float16* __restrict__ h) {
    __shared__ float red0[256];
    __shared__ float red1[256];
    int row = blockIdx.x;
    int tid = threadIdx.x;
    const float* xr = x + (size_t)row * D_MODEL;
    float v[4];
    float s = 0.0f, ss = 0.0f;
#pragma unroll
    for (int i = 0; i < 4; ++i) { v[i] = xr[tid + i * 256]; s += v[i]; ss += v[i] * v[i]; }
    red0[tid] = s; red1[tid] = ss;
    __syncthreads();
    for (int off = 128; off > 0; off >>= 1) {
        if (tid < off) { red0[tid] += red0[tid + off]; red1[tid] += red1[tid + off]; }
        __syncthreads();
    }
    float mu  = red0[0] * (1.0f / D_MODEL);
    float var = red1[0] * (1.0f / D_MODEL) - mu * mu;
    float inv = rsqrtf(var + LN_EPS);
    _Float16* hr = h + (size_t)row * D_MODEL;
#pragma unroll
    for (int i = 0; i < 4; ++i) {
        int d = tid + i * 256;
        hr[d] = (_Float16)(((v[i] - mu) * inv) * w[d] + b[d]);
    }
}

// ---------------------------------------------------------------------------
// TDM helper: async-load one 64-row x 128-col f16 slice of Wcat into LDS.
// D# group0: [0]=count=1, [1]=lds byte addr, [2]=global addr lo,
//            [3]=global addr[56:32] | type=2<<30.
// D# group1: data_size=1 (2B), tensor = 1024 x 64, tile = 128 x 64,
//            tensor_dim0_stride = 1024 elements. Tile rows pack contiguously
//            in LDS (row stride 256B). Tracked by TENSORcnt.
// ---------------------------------------------------------------------------
__device__ __forceinline__ void tdm_load_wcat_slice(const _Float16* gsrc, unsigned lds_byte_addr) {
    unsigned long long ga = (unsigned long long)(uintptr_t)gsrc;
    u32x4 g0;
    g0[0] = 1u;                                              // count=1, user descriptor
    g0[1] = lds_byte_addr;                                   // lds_addr
    g0[2] = (unsigned)(ga & 0xFFFFFFFFu);                    // global_addr[31:0]
    g0[3] = (unsigned)((ga >> 32) & 0x1FFFFFFu) | (2u << 30); // global_addr[56:32] | type=2
    u32x8 g1;
    g1[0] = (1u << 16);                                      // data_size=1 (2 bytes/elem)
    g1[1] = (1024u & 0xFFFFu) << 16;                         // tensor_dim0=1024 (lo16)
    g1[2] = (1024u >> 16) | (64u << 16);                     // tensor_dim0 hi | tensor_dim1=64 lo
    g1[3] = (0u) | ((unsigned)KCHUNK << 16);                 // tensor_dim1 hi | tile_dim0=128
    g1[4] = 64u;                                             // tile_dim1=64, tile_dim2=0
    g1[5] = 1024u;                                           // tensor_dim0_stride = 1024 elems
    g1[6] = 0u;                                              // stride0 hi | stride1 lo
    g1[7] = 0u;                                              // stride1 hi
    asm volatile("tensor_load_to_lds %0, %1" : : "s"(g0), "s"(g1) : "memory");
}

// ---------------------------------------------------------------------------
// Kernel 2: fused projections via WMMA + TDM-staged weights.
// Block = 8 waves = 8 token tiles. K loop: 8 chunks of 128, double-buffered
// Wcat slices in LDS via tensor_load_to_lds (wave 0 issues, s_wait_tensorcnt).
// Per chunk each wave runs 4 sub-K x 4 N-tile v_wmma_f32_16x16x32_f16.
// Epilogue through LDS: rw = softmax(rq+brq); us = softmax(wq+bwq)*sigmoid(wg+bwg);
// wv = wv + bwv.
// ---------------------------------------------------------------------------
__global__ __launch_bounds__(256) void proj_kernel(
        const _Float16* __restrict__ h, const _Float16* __restrict__ Wcat,
        const float* __restrict__ brq, const float* __restrict__ bwq,
        const float* __restrict__ bwv, const float* __restrict__ bwg,
        float* __restrict__ rw_g, float* __restrict__ us_g, float* __restrict__ wv_g) {
    __shared__ _Float16 wlds[2][64][KCHUNK];   // 2 x 16KB staged weight slices
    __shared__ float    cstage[8][16][64];     // 32KB epilogue staging
    int wid  = threadIdx.x >> 5;
    int lane = threadIdx.x & 31;
    int half = lane >> 4;
    int l16  = lane & 15;
    int tile = blockIdx.x * 8 + wid;
    int arow = tile * 16 + l16;

    unsigned ldsA = (unsigned)(uintptr_t)(void*)&wlds[0][0][0];
    unsigned ldsB = (unsigned)(uintptr_t)(void*)&wlds[1][0][0];

    v8f acc[4];
#pragma unroll
    for (int nt = 0; nt < 4; ++nt) acc[nt] = (v8f){};

    // Preload chunk 0 into buffer 0 (wave 0 issues the DMA for the block).
    if (wid == 0) {
        tdm_load_wcat_slice(Wcat, ldsA);
        __builtin_amdgcn_s_wait_tensorcnt(0);
    }
    __syncthreads();

    const _Float16* hrow = h + (size_t)arow * D_MODEL;
    for (int kb = 0; kb < D_MODEL / KCHUNK; ++kb) {
        // Kick off next slice into the other buffer while computing this one.
        if (wid == 0 && kb < (D_MODEL / KCHUNK) - 1)
            tdm_load_wcat_slice(Wcat + (kb + 1) * KCHUNK, (kb & 1) ? ldsA : ldsB);

        const _Float16 (*wbuf)[KCHUNK] = wlds[kb & 1];
#pragma unroll
        for (int sub = 0; sub < 4; ++sub) {
            int k0 = kb * KCHUNK + sub * 32;
            // A fragment: lanes 0-15 row M=l16 hold K = k0+{0..15};
            // lanes 16-31 hold K = k0+{16..31}.
            v16h a = *(const v16h*)(hrow + k0 + half * 16);
            int klocal = sub * 32 + half * 16;
#pragma unroll
            for (int nt = 0; nt < 4; ++nt) {
                // B fragment from LDS: lane = column N (Wcat row), 16 contig K
                v16h bf = *(const v16h*)&wbuf[nt * 16 + l16][klocal];
                acc[nt] = __builtin_amdgcn_wmma_f32_16x16x32_f16(
                    false, a, false, bf, (short)0, acc[nt], false, false);
            }
        }
        // Ensure the prefetched slice has landed before anyone reads it.
        if (wid == 0 && kb < (D_MODEL / KCHUNK) - 1)
            __builtin_amdgcn_s_wait_tensorcnt(0);
        __syncthreads();
    }

    // C layout -> LDS: element (m,n): vgpr j=m%8, lane = n + 16*(m/8)
#pragma unroll
    for (int nt = 0; nt < 4; ++nt)
#pragma unroll
        for (int j = 0; j < 8; ++j)
            cstage[wid][j + 8 * half][nt * 16 + l16] = acc[nt][j];
    __syncthreads();

    if (lane < 16) {
        int token = tile * 16 + l16;
        const float* c = &cstage[wid][l16][0];
        // read-query softmax
        float rv[8], m = -1e30f;
#pragma unroll
        for (int r = 0; r < 8; ++r) { rv[r] = c[r] + brq[r]; m = fmaxf(m, rv[r]); }
        float sum = 0.0f;
#pragma unroll
        for (int r = 0; r < 8; ++r) { rv[r] = expf(rv[r] - m); sum += rv[r]; }
        float invs = 1.0f / sum;
        // write-query softmax
        float wq[8]; m = -1e30f;
#pragma unroll
        for (int r = 0; r < 8; ++r) { wq[r] = c[8 + r] + bwq[r]; m = fmaxf(m, wq[r]); }
        float sumw = 0.0f;
#pragma unroll
        for (int r = 0; r < 8; ++r) { wq[r] = expf(wq[r] - m); sumw += wq[r]; }
        float invw = 1.0f / sumw;
        float g = 1.0f / (1.0f + expf(-(c[48] + bwg[0])));
#pragma unroll
        for (int r = 0; r < 8; ++r) {
            rw_g[(size_t)token * 8 + r] = rv[r] * invs;
            us_g[(size_t)token * 8 + r] = wq[r] * invw * g;  // fused ww*wg
        }
#pragma unroll
        for (int k = 0; k < 32; ++k)
            wv_g[(size_t)token * 32 + k] = c[16 + k] + bwv[k];
    }
}

// ---------------------------------------------------------------------------
// Kernel 3: sequential scan. One wave per batch; lane = register column d
// (D_REG == wave32). regs[r] live in 8 VGPRs; rv = sum_r rw[r]*regs[r]
// (pre-update), then regs[r] += us[r]*(wv - regs[r]).
// ---------------------------------------------------------------------------
__global__ __launch_bounds__(32) void scan_kernel(const float* __restrict__ rw_g,
                                                  const float* __restrict__ us_g,
                                                  const float* __restrict__ wv_g,
                                                  _Float16* __restrict__ rvs) {
    int b = blockIdx.x;
    int lane = threadIdx.x;
    float regs[8];
#pragma unroll
    for (int r = 0; r < 8; ++r) regs[r] = 0.0f;
    size_t base = (size_t)b * SEQL;
    for (int t = 0; t < SEQL; ++t) {
        size_t idx = base + t;
        float wvv = wv_g[idx * 32 + lane];
        float rv = 0.0f;
#pragma unroll
        for (int r = 0; r < 8; ++r) {
            float rwr = rw_g[idx * 8 + r];   // uniform -> scalar loads
            float usr = us_g[idx * 8 + r];
            rv += rwr * regs[r];
            regs[r] += usr * (wvv - regs[r]);
        }
        rvs[idx * 32 + lane] = (_Float16)rv;
    }
}

// ---------------------------------------------------------------------------
// Kernel 4: out = mix * (rvs @ Wrp^T + brp) via WMMA, K=32 single chunk.
// One block per 16-token tile; each wave covers 8 of the 64 N-tiles.
// ---------------------------------------------------------------------------
__global__ __launch_bounds__(256) void outproj_kernel(const _Float16* __restrict__ rvs,
                                                      const _Float16* __restrict__ Wrp16,
                                                      const float* __restrict__ brp,
                                                      const float* __restrict__ mix_p,
                                                      float* __restrict__ out) {
    int wid  = threadIdx.x >> 5;
    int lane = threadIdx.x & 31;
    int half = lane >> 4;
    int l16  = lane & 15;
    int tile = blockIdx.x;
    int arow = tile * 16 + l16;
    float mix = mix_p[0];

    v16h a = *(const v16h*)(rvs + (size_t)arow * 32 + half * 16);

    for (int nb = wid; nb < 64; nb += 8) {
        int dbase = nb * 16;
        v16h bf = *(const v16h*)(Wrp16 + (size_t)(dbase + l16) * 32 + half * 16);
        v8f acc = (v8f){};
        acc = __builtin_amdgcn_wmma_f32_16x16x32_f16(
            false, a, false, bf, (short)0, acc, false, false);
        int d = dbase + l16;
        float bias = brp[d];
#pragma unroll
        for (int j = 0; j < 8; ++j) {
            int m = j + 8 * half;
            out[(size_t)(tile * 16 + m) * D_MODEL + d] = mix * (acc[j] + bias);
        }
    }
}

// ---------------------------------------------------------------------------
// Launch. Workspace layout (bytes, all 256B aligned):
//   h16   @ 0          : 16384*1024*2 = 33,554,432
//   Wcat  @ 33,554,432 : 64*1024*2    =    131,072
//   Wrp16 @ 33,685,504 : 1024*32*2    =     65,536
//   rw    @ 33,751,040 : 16384*8*4    =    524,288
//   us    @ 34,275,328 : 16384*8*4    =    524,288
//   wv    @ 34,799,616 : 16384*32*4   =  2,097,152
//   rvs   @ 36,896,768 : 16384*32*2   =  1,048,576   (total ~37.9 MB)
// ---------------------------------------------------------------------------
extern "C" void kernel_launch(void* const* d_in, const int* in_sizes, int n_in,
                              void* d_out, int out_size, void* d_ws, size_t ws_size,
                              hipStream_t stream) {
    const float* x    = (const float*)d_in[0];
    const float* ln_w = (const float*)d_in[1];
    const float* ln_b = (const float*)d_in[2];
    const float* Wrq  = (const float*)d_in[3];
    const float* brq  = (const float*)d_in[4];
    const float* Wrp  = (const float*)d_in[5];
    const float* brp  = (const float*)d_in[6];
    const float* Wwq  = (const float*)d_in[7];
    const float* bwq  = (const float*)d_in[8];
    const float* Wwv  = (const float*)d_in[9];
    const float* bwv  = (const float*)d_in[10];
    const float* Wwg  = (const float*)d_in[11];
    const float* bwg  = (const float*)d_in[12];
    const float* mix  = (const float*)d_in[13];

    char* ws = (char*)d_ws;
    _Float16* h16   = (_Float16*)(ws);
    _Float16* Wcat  = (_Float16*)(ws + 33554432);
    _Float16* Wrp16 = (_Float16*)(ws + 33685504);
    float*    rw_g  = (float*)(ws + 33751040);
    float*    us_g  = (float*)(ws + 34275328);
    float*    wv_g  = (float*)(ws + 34799616);
    _Float16* rvs   = (_Float16*)(ws + 36896768);

    prep_weights<<<384, 256, 0, stream>>>(Wrq, Wwq, Wwv, Wwg, Wrp, Wcat, Wrp16);
    ln_kernel<<<NTOK, 256, 0, stream>>>(x, ln_w, ln_b, h16);
    proj_kernel<<<NTOK / 16 / 8, 256, 0, stream>>>(h16, Wcat, brq, bwq, bwv, bwg,
                                                   rw_g, us_g, wv_g);
    scan_kernel<<<BATCH, 32, 0, stream>>>(rw_g, us_g, wv_g, rvs);
    outproj_kernel<<<NTOK / 16, 256, 0, stream>>>(rvs, Wrp16, brp, mix, (float*)d_out);
}
